// Emphasize_83339545411892
// MI455X (gfx1250) — matrix operand order repaired and checked by reading, not compiled, not run on hardware
//
#include <hip/hip_runtime.h>
#include <hip/hip_bf16.h>

// ---------------------------------------------------------------------------
// Emphasize: out = real(ifft2(ifftshift(fftshift(fft2(x)) * mask)))
// Tensor-core DFT with a self-transposing stage operator:
//   s(M) = F @ M^T   =>  s(s(x)) = F x F = fft2(x)   (F symmetric, F^T = F)
//   t(M) = F* @ M^T  =>  t(t(Y)) = F* Y F* = S^2 * ifft2(Y)
// Every stage: A = Fr/Fi (fixed, L2-resident), B = previous output read
// along its rows => ALL operand staging is contiguous 16B async DMA to LDS,
// pre-swizzled into WMMA fragment order. Double-buffered on ASYNCcnt.
// ---------------------------------------------------------------------------

typedef __attribute__((ext_vector_type(16))) __bf16        v16bf;
typedef __attribute__((ext_vector_type(4)))  __bf16        v4bf;
typedef __attribute__((ext_vector_type(8)))  float         v8f;
typedef __attribute__((ext_vector_type(8)))  unsigned int  v8u;

static constexpr int SZ = 512;   // image / DFT size
static constexpr int NB = 64;    // batch
static constexpr int NC = 64;    // clusters

// ---- CDNA5 helpers ---------------------------------------------------------

static __device__ inline v8f wmma_bf16(v16bf a, v16bf b, v8f c) {
  return __builtin_amdgcn_wmma_f32_16x16x32_bf16(
      false, a, false, b, (short)0, c, false, false);
}

static __device__ inline v16bf bneg(v16bf x) {  // negate 16 packed bf16
  v8u u;
  __builtin_memcpy(&u, &x, sizeof(u));
  u ^= 0x80008000u;
  v16bf r;
  __builtin_memcpy(&r, &u, sizeof(r));
  return r;
}

// async DMA: 16 bytes global -> LDS, per-lane swizzled LDS destination.
static __device__ inline void cp_async16(const __bf16* gsrc, __bf16* lds) {
  unsigned lds_off = (unsigned)(unsigned long long)lds;  // low 32b = LDS addr
  asm volatile("global_load_async_to_lds_b128 %0, %1, off"
               :: "v"(lds_off), "v"(gsrc) : "memory");
}
template <int N>
static __device__ inline void wait_async() {
  asm volatile("s_wait_asynccnt %0" :: "n"(N) : "memory");
}

// ---- small prep kernels ----------------------------------------------------

__global__ void k_twiddle(__bf16* __restrict__ Fr, __bf16* __restrict__ Fi) {
  int i = blockIdx.x * 256 + threadIdx.x;          // 0 .. S*S-1
  int k = i >> 9, n = i & (SZ - 1);
  int m = (k * n) & (SZ - 1);                      // (k*n) mod 512, exact
  float ang = (float)m * (-6.283185307179586f / (float)SZ);
  float s, c;
  __sincosf(ang, &s, &c);
  Fr[i] = (__bf16)c;
  Fi[i] = (__bf16)s;
}

__global__ void k_cvt(const float* __restrict__ x, __bf16* __restrict__ xb) {
  int i = (blockIdx.x * 256 + threadIdx.x) * 4;
  float4 v = *(const float4*)(x + i);
  v4bf o;
  o[0] = (__bf16)v.x; o[1] = (__bf16)v.y; o[2] = (__bf16)v.z; o[3] = (__bf16)v.w;
  *(v4bf*)(xb + i) = o;
}

__global__ void k_cid_init(int* __restrict__ cid) {
  cid[blockIdx.x * 256 + threadIdx.x] = -1;
}

// fftshift folded into the scatter coordinates.
__global__ void k_cid_scatter(int* __restrict__ cid,
                              const int* __restrict__ ix,
                              const int* __restrict__ iy,
                              const int* __restrict__ cl, int P) {
  int p = blockIdx.x * 256 + threadIdx.x;
  if (p >= P) return;
  int u = (ix[p] + SZ / 2) & (SZ - 1);
  int v = (iy[p] + SZ / 2) & (SZ - 1);
  cid[u * SZ + v] = cl[p];
}

// ---- fused complex GEMM stages (all compute  Out = {F or F*} @ In^T) ------
// MODE 0: Tr = Fr@x^T          Ti = Fi@x^T             (single real B)
// MODE 1: Yr = Fr@Tr^T-Fi@Ti^T Yi = Fr@Ti^T+Fi@Tr^T    * mask (epilogue)
// MODE 2: Ar = Fr@Yr^T+Fi@Yi^T Ai = Fr@Yi^T-Fi@Yr^T
// MODE 3: out = (Fr@Ar^T + Fi@Ai^T) / S^2              (f32 output)
//
// Block tile 128(M) x 64(N), K-step 32, double-buffered LDS.
// 8 waves (wave32) in a 4x2 grid; wave tile 32x32 = 2x2 WMMA per component.

template <int MODE>
__global__ __launch_bounds__(256)
void k_gemm(const __bf16* __restrict__ Frm, const __bf16* __restrict__ Fim,
            const __bf16* __restrict__ B1, const __bf16* __restrict__ B2,
            void* __restrict__ O1p, void* __restrict__ O2p,
            const int* __restrict__ cid, const float* __restrict__ pimp) {
  constexpr bool HAS_B2 = (MODE != 0);
  constexpr int  NSTG   = HAS_B2 ? 6 : 5;   // async ops per wave per stage

  // A: 16x32 bf16 fragment order (ISA 7.12.2): lane = m + 16*((k>>3)&1),
  //    elem = (k&7) + 8*(k>>4).   B: lane = n + 16*((k>>3)&1), elem as A.
  __shared__ __bf16 sA[2][2 * 128 * 32];   // [buf][mat q][sub s][lane][elem]
  __shared__ __bf16 sB[2][2 * 32 * 64];

  const int tid = threadIdx.x;
  const int lane = tid & 31;
  const int wave = tid >> 5;
  const int wM = wave >> 1;                // 0..3
  const int wN = wave & 1;                 // 0..1
  const int b = blockIdx.z;
  const int mBase = blockIdx.y * 128;
  const int nBase = blockIdx.x * 64;
  const size_t bo = (size_t)b * (SZ * SZ);

  const __bf16* B1p = B1 + bo;
  const __bf16* B2p = HAS_B2 ? (B2 + bo) : (const __bf16*)nullptr;

  // ---- one pipeline stage: issue NSTG async 16B copies into buffer `pb` ---
  auto stage = [&](int pb, int kt) {
    // A chunks: c = [q:1][row:7][kgrp:2], 4 per thread
#pragma unroll
    for (int i = 0; i < 4; ++i) {
      int c = tid + 256 * i;
      int q = c >> 9;
      int r = (c >> 2) & 127;
      int g = c & 3;
      const __bf16* src = (q ? Fim : Frm) + (size_t)(mBase + r) * SZ + kt + g * 8;
      int eoff = (q << 12) + ((r >> 4) << 9) +
                 (((r & 15) | ((g & 1) << 4)) << 4) + ((g >> 1) << 3);
      cp_async16(src, &sA[pb][eoff]);
    }
    // B chunks: c = [q:1][nsub:2][lane:5][half:1], element (k,n) = In[n][k]
#pragma unroll
    for (int i = 0; i < 2; ++i) {
      int c = tid + 256 * i;
      int q = c >> 8;
      if (!HAS_B2 && q) break;             // uniform: whole second pass skipped
      int s = (c >> 6) & 3;
      int l = (c >> 1) & 31;
      int d = c & 1;
      const __bf16* Bp = q ? B2p : B1p;
      int nrow = nBase + s * 16 + (l & 15);
      int k0 = kt + ((l >> 4) << 3) + (d << 4);
      const __bf16* src = Bp + (size_t)nrow * SZ + k0;   // contiguous along k
      int eoff = (q << 11) + (s << 9) + (l << 4) + (d << 3);
      cp_async16(src, &sB[pb][eoff]);
    }
  };

  v8f acc1[2][2], acc2[2][2];
#pragma unroll
  for (int mi = 0; mi < 2; ++mi)
#pragma unroll
    for (int ni = 0; ni < 2; ++ni) {
      acc1[mi][ni] = (v8f)0.0f;
      acc2[mi][ni] = (v8f)0.0f;
    }

  stage(0, 0);                             // prime the pipeline
  int p = 0;
  for (int kt = 0; kt < SZ; kt += 32) {
    const bool hasNext = (kt + 32) < SZ;
    if (hasNext) stage(p ^ 1, kt + 32);    // overlap DMA with this step's WMMA
    if (hasNext) wait_async<NSTG>(); else wait_async<0>();
    __syncthreads();                       // buffer p complete for all waves

    v16bf a1f[2], a2f[2], b1f[2], b2f[2];
#pragma unroll
    for (int mi = 0; mi < 2; ++mi) {
      int s = wM * 2 + mi;
      a1f[mi] = *(const v16bf*)(&sA[p][(s << 9) + (lane << 4)]);
      a2f[mi] = *(const v16bf*)(&sA[p][4096 + (s << 9) + (lane << 4)]);
    }
#pragma unroll
    for (int ni = 0; ni < 2; ++ni) {
      int s = wN * 2 + ni;
      b1f[ni] = *(const v16bf*)(&sB[p][(s << 9) + (lane << 4)]);
      if (HAS_B2) b2f[ni] = *(const v16bf*)(&sB[p][2048 + (s << 9) + (lane << 4)]);
    }

#pragma unroll
    for (int mi = 0; mi < 2; ++mi) {
      v16bf a2n = (MODE == 1 || MODE == 2) ? bneg(a2f[mi]) : a2f[mi];
#pragma unroll
      for (int ni = 0; ni < 2; ++ni) {
        acc1[mi][ni] = wmma_bf16(a1f[mi], b1f[ni], acc1[mi][ni]);
        if (MODE == 0) {
          acc2[mi][ni] = wmma_bf16(a2f[mi], b1f[ni], acc2[mi][ni]);
        } else if (MODE == 1) {
          acc1[mi][ni] = wmma_bf16(a2n,     b2f[ni], acc1[mi][ni]);  // -Fi@Ti^T
          acc2[mi][ni] = wmma_bf16(a1f[mi], b2f[ni], acc2[mi][ni]);
          acc2[mi][ni] = wmma_bf16(a2f[mi], b1f[ni], acc2[mi][ni]);
        } else if (MODE == 2) {
          acc1[mi][ni] = wmma_bf16(a2f[mi], b2f[ni], acc1[mi][ni]);
          acc2[mi][ni] = wmma_bf16(a1f[mi], b2f[ni], acc2[mi][ni]);
          acc2[mi][ni] = wmma_bf16(a2n,     b1f[ni], acc2[mi][ni]);  // -Fi@Yr^T
        } else {
          acc1[mi][ni] = wmma_bf16(a2f[mi], b2f[ni], acc1[mi][ni]);
        }
      }
    }
    __syncthreads();                       // all reads of buf p done
    p ^= 1;
  }

  // ---- epilogue: D layout: N = lane&15, M = vgpr + 8*(lane>>4) -------------
  const int h = lane >> 4;
  const int nn = lane & 15;
#pragma unroll
  for (int mi = 0; mi < 2; ++mi) {
#pragma unroll
    for (int ni = 0; ni < 2; ++ni) {
      int rb = mBase + wM * 32 + mi * 16 + h * 8;
      int cc = nBase + wN * 32 + ni * 16 + nn;
#pragma unroll
      for (int r = 0; r < 8; ++r) {
        size_t idx = (size_t)(rb + r) * SZ + cc;
        float v1 = acc1[mi][ni][r];
        if (MODE == 3) {
          ((float*)O1p)[bo + idx] = v1 * (1.0f / 262144.0f);  // 1/S^2
        } else {
          float v2 = acc2[mi][ni][r];
          if (MODE == 1) {  // fused frequency mask (fftshift folded in)
            int cd = cid[idx];
            float m = (cd < 0) ? 1.0f : pimp[b * NC + cd];
            v1 *= m;
            v2 *= m;
          }
          ((__bf16*)O1p)[bo + idx] = (__bf16)v1;
          ((__bf16*)O2p)[bo + idx] = (__bf16)v2;
        }
      }
    }
  }
}

// ---------------------------------------------------------------------------

extern "C" void kernel_launch(void* const* d_in, const int* in_sizes, int n_in,
                              void* d_out, int out_size, void* d_ws, size_t ws_size,
                              hipStream_t stream) {
  const float* x    = (const float*)d_in[0];   // (64,512,512) f32
  const float* pimp = (const float*)d_in[1];   // (64,64) f32
  const int*   ixx  = (const int*)d_in[2];     // (P,) i32
  const int*   ixy  = (const int*)d_in[3];     // (P,) i32
  const int*   cl   = (const int*)d_in[4];     // (P,) i32
  float* out = (float*)d_out;
  const int P = in_sizes[2];

  char* w = (char*)d_ws;  // (MB): [0,.5)Fr [.5,1)Fi [1,2)cid [2,34)xb
                          // [34,66)Tr/Ar [66,98)Ti/Ai [98,130)Yr [130,162)Yi
  __bf16* Fr  = (__bf16*)(w);
  __bf16* Fi  = (__bf16*)(w + (size_t)512 * 1024);
  int*    cid = (int*)   (w + ((size_t)1 << 20));
  __bf16* xb  = (__bf16*)(w + ((size_t)2 << 20));
  __bf16* Tr  = (__bf16*)(w + ((size_t)34 << 20));
  __bf16* Ti  = (__bf16*)(w + ((size_t)66 << 20));
  __bf16* Yr  = (__bf16*)(w + ((size_t)98 << 20));
  __bf16* Yi  = (__bf16*)(w + ((size_t)130 << 20));

  k_twiddle<<<SZ * SZ / 256, 256, 0, stream>>>(Fr, Fi);
  k_cvt<<<NB * SZ * SZ / 1024, 256, 0, stream>>>(x, xb);
  k_cid_init<<<SZ * SZ / 256, 256, 0, stream>>>(cid);
  k_cid_scatter<<<(P + 255) / 256, 256, 0, stream>>>(cid, ixx, ixy, cl, P);

  dim3 grid(SZ / 64, SZ / 128, NB), blk(256);
  k_gemm<0><<<grid, blk, 0, stream>>>(Fr, Fi, xb, nullptr, Tr, Ti, nullptr, nullptr);
  k_gemm<1><<<grid, blk, 0, stream>>>(Fr, Fi, Tr, Ti, Yr, Yi, cid, pimp);
  k_gemm<2><<<grid, blk, 0, stream>>>(Fr, Fi, Yr, Yi, Tr, Ti, nullptr, nullptr);
  k_gemm<3><<<grid, blk, 0, stream>>>(Fr, Fi, Tr, Ti, out, nullptr, nullptr, nullptr);
}
